// Model_39676907885418
// MI455X (gfx1250) — matrix operand order repaired and checked by reading, not compiled, not run on hardware
//
#include <hip/hip_runtime.h>
#include <hip/hip_bf16.h>

#define S_LEN   2048
#define D_DIM   128
#define BLOCK_M 128
#define BLOCK_N 32
#define NWAVES  8
#define KTILES  (S_LEN / BLOCK_N)   // 64

// dropout constants (hash-based inverted dropout; JAX RNG cannot be matched bit-exact)
#define SEED_X  0x9E3779B9u
#define SEED_A  0xC2B2AE35u
#define TH_X    3435973836u          // 0.8 * 2^32  (keep prob for input dropout)
#define TH_A    3865470566u          // 0.9 * 2^32  (keep prob for attn dropout)
#define LOG2E   1.4426950408889634f
// (1/sqrt(128)) / 0.8 * log2(e)  -- scores come out of WMMA already in log2 domain
#define QFAC_L2 0.15939740f
#define INV_KA  1.1111111111f        // 1 / 0.9

typedef __attribute__((ext_vector_type(16))) __bf16 v16bf;
typedef __attribute__((ext_vector_type(8)))  __bf16 v8bf;
typedef __attribute__((ext_vector_type(8)))  float  v8f;

union V16U { v16bf v; v8bf h[2]; __bf16 e[16]; };

#if __has_builtin(__builtin_amdgcn_exp2f)
#define EXP2(x) __builtin_amdgcn_exp2f(x)
#else
#define EXP2(x) exp2f(x)
#endif

#if __has_builtin(__builtin_amdgcn_rcpf)
#define RCP(x) __builtin_amdgcn_rcpf(x)
#else
#define RCP(x) (1.0f / (x))
#endif

// xor-butterfly lane swaps within 16-lane halves via ds_swizzle (group-of-32 mode)
template <int IMM>
__device__ __forceinline__ float swz(float v) {
  return __int_as_float(__builtin_amdgcn_ds_swizzle(__float_as_int(v), IMM));
}
__device__ __forceinline__ float rmax16(float v) {
  v = fmaxf(v, swz<0x041F>(v));  // xor 1
  v = fmaxf(v, swz<0x081F>(v));  // xor 2
  v = fmaxf(v, swz<0x101F>(v));  // xor 4
  v = fmaxf(v, swz<0x201F>(v));  // xor 8
  return v;
}
__device__ __forceinline__ float rsum16(float v) {
  v += swz<0x041F>(v);
  v += swz<0x081F>(v);
  v += swz<0x101F>(v);
  v += swz<0x201F>(v);
  return v;
}

__device__ __forceinline__ unsigned hmix(unsigned x) {
  x ^= x >> 16; x *= 0x7feb352du;
  x ^= x >> 15; x *= 0x846ca68bu;
  x ^= x >> 16; return x;
}

__device__ __forceinline__ unsigned short f2bf_bits(float f) {
  unsigned u = __float_as_uint(f);
  u += 0x7fffu + ((u >> 16) & 1u);   // round-to-nearest-even
  return (unsigned short)(u >> 16);
}
__device__ __forceinline__ __bf16 f2bf(float f) {
  union { unsigned short s; __bf16 b; } cv;
  cv.s = f2bf_bits(f);
  return cv.b;
}

__global__ __launch_bounds__(256)
void fa_dropout_kernel(const float* __restrict__ xp, const float* __restrict__ kp,
                       const float* __restrict__ vp, const float* __restrict__ maskp,
                       float* __restrict__ outp) {
  // double-buffered K ([kv][d]) and V (transposed [d][kv]) tiles + per-wave P staging
  __shared__ __align__(32) __bf16 KtB[2][D_DIM * BLOCK_N];
  __shared__ __align__(32) __bf16 VtB[2][BLOCK_N * D_DIM];
  __shared__ __align__(32) __bf16 Pt[NWAVES * 16 * BLOCK_N];

  const int bh    = blockIdx.y;
  const int tid   = threadIdx.x;
  const int wave  = tid >> 5;
  const int lane  = tid & 31;
  const int lh    = lane & 15;       // N / M-in-half index
  const int hg    = lane >> 4;       // lane half
  const int qbase = blockIdx.x * BLOCK_M + wave * 16;

  // cooperative-load mapping for K/V tiles: 256 threads, 32 rows x 128 cols fp32
  const int rrow = tid >> 3;         // 0..31 kv row in tile
  const int c0   = (tid & 7) << 4;   // d start (16 floats per thread)
  const size_t kvrowbase = ((size_t)bh * S_LEN + rrow) * D_DIM + c0;

  // ---- Load Q once: dropout + scale*log2e folded in, packed as 4 bf16 A-fragments ----
  const int qrow_frag = qbase + lh;  // A-layout: lanes 0-15 carry M=0..15 (both halves)
  const float* xr = xp + ((size_t)bh * S_LEN + qrow_frag) * D_DIM;
  const unsigned xib = ((unsigned)bh * S_LEN + (unsigned)qrow_frag) * D_DIM;
  V16U qf[4];
#pragma unroll
  for (int c = 0; c < 4; ++c) {
#pragma unroll
    for (int g = 0; g < 2; ++g) {
      const int dstart = c * 32 + hg * 16 + g * 8;
      float t8[8];
      *(float4*)&t8[0] = *(const float4*)(xr + dstart);
      *(float4*)&t8[4] = *(const float4*)(xr + dstart + 4);
#pragma unroll
      for (int j = 0; j < 8; ++j) {
        const unsigned idx = xib + (unsigned)(dstart + j);
        const float val = (hmix(idx ^ SEED_X) < TH_X) ? t8[j] * QFAC_L2 : 0.0f;
        qf[c].e[g * 8 + j] = f2bf(val);
      }
    }
  }

  v8f vzero = {};
  v8f acc[8];
#pragma unroll
  for (int t = 0; t < 8; ++t) acc[t] = vzero;
  float mrow[8], lrow[8];
#pragma unroll
  for (int r = 0; r < 8; ++r) { mrow[r] = -3.0e38f; lrow[r] = 0.0f; }

  // in-flight tile registers for software pipelining
  float fnk[16], fnv[16];

  // ---- prologue: fetch tile 0 and stage it into buffer 0 ----
#pragma unroll
  for (int j = 0; j < 4; ++j) {
    *(float4*)&fnk[j * 4] = *(const float4*)(kp + kvrowbase + j * 4);
    *(float4*)&fnv[j * 4] = *(const float4*)(vp + kvrowbase + j * 4);
  }
  {
    __bf16* Kt = KtB[0]; __bf16* Vt = VtB[0];
#pragma unroll
    for (int j = 0; j < 16; ++j) Vt[(c0 + j) * BLOCK_N + rrow] = f2bf(fnv[j]);
    unsigned pk[8];
#pragma unroll
    for (int j = 0; j < 8; ++j)
      pk[j] = (unsigned)f2bf_bits(fnk[2 * j]) | ((unsigned)f2bf_bits(fnk[2 * j + 1]) << 16);
    uint4* vd = (uint4*)&Kt[rrow * D_DIM + c0];
    vd[0] = make_uint4(pk[0], pk[1], pk[2], pk[3]);
    vd[1] = make_uint4(pk[4], pk[5], pk[6], pk[7]);
  }

  for (int kt = 0; kt < KTILES; ++kt) {
    // ---- issue next tile's global loads NOW; consume them after the compute ----
    if (kt + 1 < KTILES) {
      const size_t gn = kvrowbase + (size_t)(kt + 1) * BLOCK_N * D_DIM;
#pragma unroll
      for (int j = 0; j < 4; ++j) {
        *(float4*)&fnk[j * 4] = *(const float4*)(kp + gn + j * 4);
        *(float4*)&fnv[j * 4] = *(const float4*)(vp + gn + j * 4);
      }
      if (kt + 2 < KTILES) {  // warm L2 for the tile after that (global_prefetch_b8)
        const size_t g2 = kvrowbase + (size_t)(kt + 2) * BLOCK_N * D_DIM;
        __builtin_prefetch(kp + g2, 0, 1);
        __builtin_prefetch(vp + g2, 0, 1);
      }
    }

    __syncthreads();   // buffer (kt&1) fully staged by all waves
    const __bf16* Kt = KtB[kt & 1];
    const __bf16* Vt = VtB[kt & 1];

    // ---- scores: S = Q x K^T in log2 domain (two 16x16 N-tiles, 4 d-chunks) ----
    v8f s0 = vzero, s1 = vzero;
#pragma unroll
    for (int c = 0; c < 4; ++c) {
      v16bf b0 = *(const v16bf*)&Kt[lh * D_DIM + c * 32 + hg * 16];
      v16bf b1 = *(const v16bf*)&Kt[(16 + lh) * D_DIM + c * 32 + hg * 16];
      s0 = __builtin_amdgcn_wmma_f32_16x16x32_bf16(false, qf[c].v, false, b0,
                                                   (short)0, s0, false, false);
      s1 = __builtin_amdgcn_wmma_f32_16x16x32_bf16(false, qf[c].v, false, b1,
                                                   (short)0, s1, false, false);
    }

    // ---- mask (x log2e) + online softmax (exp2) + attn dropout, P -> LDS ----
    const int kbase = kt * BLOCK_N;
    const unsigned pbase = (unsigned)wave * 16 * BLOCK_N;
#pragma unroll
    for (int r = 0; r < 8; ++r) {
      const int qrow = qbase + r + hg * 8;           // C-layout row for VGPR r
      const float m0 = maskp[(size_t)qrow * S_LEN + kbase + lh];
      const float m1 = maskp[(size_t)qrow * S_LEN + kbase + 16 + lh];
      const float v0 = fmaf(m0, LOG2E, s0[r]);
      const float v1 = fmaf(m1, LOG2E, s1[r]);
      const float mx = rmax16(fmaxf(v0, v1));
      const float mnew  = fmaxf(mrow[r], mx);
      const float alpha = EXP2(mrow[r] - mnew);
      mrow[r] = mnew;
      float p0 = EXP2(v0 - mnew);
      float p1 = EXP2(v1 - mnew);
      const float rs = rsum16(p0 + p1);              // denominator BEFORE dropout
      lrow[r] = lrow[r] * alpha + rs;
      const unsigned qg = (unsigned)bh * S_LEN + (unsigned)qrow;
      const unsigned i0 = qg * S_LEN + (unsigned)(kbase + lh);
      p0 = (hmix(i0 ^ SEED_A)         < TH_A) ? p0 * INV_KA : 0.0f;
      p1 = (hmix((i0 + 16u) ^ SEED_A) < TH_A) ? p1 * INV_KA : 0.0f;
#pragma unroll
      for (int t = 0; t < 8; ++t) acc[t][r] *= alpha;
      const int mr = r + hg * 8;
      Pt[pbase + mr * BLOCK_N + lh]      = f2bf(p0);
      Pt[pbase + mr * BLOCK_N + 16 + lh] = f2bf(p1);
    }

    // ---- re-layout P (C-layout -> A-layout) via wave-private LDS, then P x V ----
    V16U pa;
    pa.h[0] = *(const v8bf*)&Pt[pbase + lh * BLOCK_N + hg * 16];
    pa.h[1] = *(const v8bf*)&Pt[pbase + lh * BLOCK_N + hg * 16 + 8];
#pragma unroll
    for (int t = 0; t < 8; ++t) {
      v16bf vb = *(const v16bf*)&Vt[(t * 16 + lh) * BLOCK_N + hg * 16];
      acc[t] = __builtin_amdgcn_wmma_f32_16x16x32_bf16(false, pa.v, false, vb,
                                                       (short)0, acc[t], false, false);
    }

    // ---- drain the in-flight tile: convert + stage into the other buffer ----
    if (kt + 1 < KTILES) {
      __bf16* Ktn = KtB[(kt + 1) & 1];
      __bf16* Vtn = VtB[(kt + 1) & 1];
#pragma unroll
      for (int j = 0; j < 16; ++j) Vtn[(c0 + j) * BLOCK_N + rrow] = f2bf(fnv[j]);
      unsigned pk[8];
#pragma unroll
      for (int j = 0; j < 8; ++j)
        pk[j] = (unsigned)f2bf_bits(fnk[2 * j]) | ((unsigned)f2bf_bits(fnk[2 * j + 1]) << 16);
      uint4* vd = (uint4*)&Ktn[rrow * D_DIM + c0];
      vd[0] = make_uint4(pk[0], pk[1], pk[2], pk[3]);
      vd[1] = make_uint4(pk[4], pk[5], pk[6], pk[7]);
    }
  }

  // ---- epilogue: normalize (one rcp per row) and store fp32 ----
#pragma unroll
  for (int r = 0; r < 8; ++r) {
    const int qrow = qbase + r + hg * 8;
    const float inv = RCP(lrow[r]);
    float* orow = outp + ((size_t)bh * S_LEN + qrow) * D_DIM;
#pragma unroll
    for (int t = 0; t < 8; ++t) {
      orow[t * 16 + lh] = acc[t][r] * inv;
    }
  }
}

extern "C" void kernel_launch(void* const* d_in, const int* in_sizes, int n_in,
                              void* d_out, int out_size, void* d_ws, size_t ws_size,
                              hipStream_t stream) {
  const float* x = (const float*)d_in[0];
  const float* k = (const float*)d_in[1];
  const float* v = (const float*)d_in[2];
  const float* m = (const float*)d_in[3];
  float* out = (float*)d_out;
  const int bh = in_sizes[0] / (S_LEN * D_DIM);   // B*H = 32
  dim3 grid(S_LEN / BLOCK_M, bh);
  fa_dropout_kernel<<<grid, 256, 0, stream>>>(x, k, v, m, out);
}
